// Stage0_10574209483486
// MI455X (gfx1250) — matrix operand order repaired and checked by reading, compiled, not run on hardware
//
#include <hip/hip_runtime.h>
#include <hip/hip_bf16.h>
#include <math.h>

typedef __attribute__((ext_vector_type(16))) _Float16 v16h;
typedef __attribute__((ext_vector_type(8)))  float    v8f;

#define NVOX 262144   // N = input active voxels
#define MVOX 131072   // M = pooled voxels
#define EPSN 1e-5f

// ---------------------------------------------------------------------------
// Stage 1: conv1  (Cin=3, Cout=64, 27 taps) -- VALU, one block per voxel.
// ---------------------------------------------------------------------------
__global__ void conv1_kernel(const float* __restrict__ feats,
                             const float* __restrict__ W1,
                             const int*   __restrict__ nbr1,
                             float*       __restrict__ y1, int rows) {
    __shared__ float lf[27 * 4];
    const int voxel = blockIdx.x;
    const int co    = threadIdx.x;          // 64 threads = 64 output channels
    if (co < 27) {
        int idx = nbr1[(size_t)co * rows + voxel];
        float a = 0.f, b = 0.f, c = 0.f;
        if (idx >= 0) {
            const float* f = feats + (size_t)idx * 3;
            a = f[0]; b = f[1]; c = f[2];
        }
        lf[co * 4 + 0] = a; lf[co * 4 + 1] = b; lf[co * 4 + 2] = c;
    }
    __syncthreads();
    float acc = 0.f;
    #pragma unroll
    for (int k = 0; k < 27; ++k) {
        const float* w = W1 + (size_t)k * 3 * 64 + co;
        acc += lf[k * 4 + 0] * w[0] + lf[k * 4 + 1] * w[64] + lf[k * 4 + 2] * w[128];
    }
    y1[(size_t)voxel * 64 + co] = acc;
}

// ---------------------------------------------------------------------------
// Deterministic per-channel stats: partial sums per block, fixed-order final.
// ---------------------------------------------------------------------------
__global__ void stats_partial(const float* __restrict__ x, int rows, int C,
                              float* __restrict__ partials) {
    const int tid = threadIdx.x;            // 256
    const int ch  = tid & (C - 1);
    const int rl  = tid / C;
    const int rpb = 256 / C;
    float s = 0.f, q = 0.f;
    for (long r = (long)blockIdx.x * rpb + rl; r < rows; r += (long)gridDim.x * rpb) {
        float v = x[r * C + ch];
        s += v; q += v * v;
    }
    __shared__ float ls[256], lq[256];
    ls[tid] = s; lq[tid] = q;
    __syncthreads();
    for (int off = 128; off >= C; off >>= 1) {
        if (tid < off) { ls[tid] += ls[tid + off]; lq[tid] += lq[tid + off]; }
        __syncthreads();
    }
    if (tid < C) {
        partials[(size_t)blockIdx.x * 2 * C + tid]     = ls[tid];
        partials[(size_t)blockIdx.x * 2 * C + C + tid] = lq[tid];
    }
}

__global__ void stats_final(const float* __restrict__ partials, int C, int nblk,
                            float* __restrict__ stats) {
    const int tid = threadIdx.x;            // 256
    if (tid < 2 * C) {
        float s = 0.f;
        for (int b = 0; b < nblk; ++b) s += partials[(size_t)b * 2 * C + tid];
        stats[tid] = s;
    }
}

// ---------------------------------------------------------------------------
// Normalize + ReLU -> f16 activations.
// ---------------------------------------------------------------------------
__global__ void normrelu_f16(const float* __restrict__ x, _Float16* __restrict__ y,
                             long total, int Cm1, const float* __restrict__ stats,
                             float invRows) {
    for (long i = (long)blockIdx.x * blockDim.x + threadIdx.x; i < total;
         i += (long)gridDim.x * blockDim.x) {
        int ch    = (int)(i & Cm1);
        float mu  = stats[ch] * invRows;
        float var = stats[Cm1 + 1 + ch] * invRows - mu * mu;
        float r   = rsqrtf(var + EPSN);
        float v   = (x[i] - mu) * r;
        v = v > 0.f ? v : 0.f;
        y[i] = (_Float16)v;
    }
}

// In-place final normalize + ReLU (f32, C=128) on d_out.
__global__ void normrelu_f32(float* __restrict__ x, long total, int Cm1,
                             const float* __restrict__ stats, float invRows) {
    for (long i = (long)blockIdx.x * blockDim.x + threadIdx.x; i < total;
         i += (long)gridDim.x * blockDim.x) {
        int ch    = (int)(i & Cm1);
        float mu  = stats[ch] * invRows;
        float var = stats[Cm1 + 1 + ch] * invRows - mu * mu;
        float r   = rsqrtf(var + EPSN);
        float v   = (x[i] - mu) * r;
        x[i] = v > 0.f ? v : 0.f;
    }
}

// ---------------------------------------------------------------------------
// Max-pool raw conv2 output over 8 child offsets, then norm+ReLU (monotone,
// commutes with max) -> f16 pooled activations.
// ---------------------------------------------------------------------------
__global__ void pool_normrelu(const float* __restrict__ y2,
                              const int* __restrict__ pool_map,
                              const float* __restrict__ stats,
                              _Float16* __restrict__ p, int M, float invRows) {
    long i = (long)blockIdx.x * blockDim.x + threadIdx.x;
    if (i >= (long)M * 64) return;
    int  ch = (int)(i & 63);
    long m  = i >> 6;
    float best = -INFINITY;
    #pragma unroll
    for (int o = 0; o < 8; ++o) {
        int idx = pool_map[(size_t)o * M + m];
        if (idx >= 0) best = fmaxf(best, y2[(size_t)idx * 64 + ch]);
    }
    float mu  = stats[ch] * invRows;
    float var = stats[64 + ch] * invRows - mu * mu;
    float r   = rsqrtf(var + EPSN);
    float v   = (best - mu) * r;
    v = fmaxf(v, 0.f);
    p[i] = (_Float16)v;
}

// ---------------------------------------------------------------------------
// Pack W[k][cin][cout] (f32) into per-lane-contiguous f16 B fragments.
// Fragment (k, c, nt) is the 32x16 B tile: cin = c*32 + kk, cout = nt*16 + n.
// B layout (16-bit, 32x16): lane<16 -> col n=lane, K={0..7,16..23} in 16 halves;
//                           lane>=16 -> col n=lane-16, K={8..15,24..31}.
// bp[((k*2+c)*NT+nt)*512 + lane*16 + t]
// ---------------------------------------------------------------------------
__global__ void pack_w(const float* __restrict__ W, _Float16* __restrict__ bp,
                       int Cin, int Cout, int NT, int total) {
    int id = blockIdx.x * 256 + threadIdx.x;
    if (id >= total) return;
    int t    = id & 15;
    int lane = (id >> 4) & 31;
    int rest = id >> 9;                 // (k*2+c)*NT + nt
    int nt   = rest % NT;
    int kc   = rest / NT;
    int c    = kc & 1;
    int k    = kc >> 1;
    int kk   = c * 32 + ((lane >= 16) ? 8 : 0) + t + ((t >= 8) ? 8 : 0);
    int col  = nt * 16 + (lane & 15);
    bp[id] = (_Float16)W[((size_t)k * Cin + kk) * Cout + col];
}

// ---------------------------------------------------------------------------
// WMMA gather-GEMM sparse conv: Cin=64, Cout = NT*16, 27 taps.
// 256 threads = 8 waves; one wave32 per 16-row output tile.
//
// B fragments are staged cooperatively through LDS, one (k,c) step at a time,
// double-buffered: global->VGPR loads for step s+1 are issued before the
// WMMAs of step s, then committed to the alternate LDS buffer behind a
// barrier, so the VMEM latency hides under compute. All 8 waves share one
// staging (8x less VMEM B-traffic; fragment reads move to the DS pipe).
//
// A fragment per ISA 16-bit 16x32 layout:
//   lane<16 : row M=lane,    halves K = {c*32+0..7,  c*32+16..23}
//   lane>=16: row M=lane-16, halves K = {c*32+8..15, c*32+24..31}
// Invalid neighbors masked to zero (no divergence at the WMMA).
// ---------------------------------------------------------------------------
template <int NT>
__global__ void conv_wmma(const _Float16* __restrict__ xin,
                          const _Float16* __restrict__ bpack,
                          const int* __restrict__ nbr,
                          float* __restrict__ out, int rows) {
    constexpr int STEP_HALVES = NT * 512;            // one (k,c) step of B
    constexpr int U4_PER_THR  = (STEP_HALVES * 2) / 16 / 256;   // NT/4
    __shared__ alignas(32) _Float16 ldsb[2][STEP_HALVES];

    const int tid  = threadIdx.x;
    const int lane = tid & 31;
    const int tile = blockIdx.x * 8 + (tid >> 5);
    const int row0 = tile * 16;
    const int m    = lane & 15;
    const bool hi  = lane >= 16;

    v8f acc[NT];
    v8f zero = {0.f, 0.f, 0.f, 0.f, 0.f, 0.f, 0.f, 0.f};
    #pragma unroll
    for (int nt = 0; nt < NT; ++nt) acc[nt] = zero;

    // Prologue: stage step 0 into LDS buffer 0.
    uint4 stage[U4_PER_THR];
    {
        const uint4* g = (const uint4*)bpack;
        #pragma unroll
        for (int j = 0; j < U4_PER_THR; ++j) stage[j] = g[tid + j * 256];
        uint4* l = (uint4*)ldsb[0];
        #pragma unroll
        for (int j = 0; j < U4_PER_THR; ++j) l[tid + j * 256] = stage[j];
    }
    __syncthreads();

    for (int s = 0; s < 54; ++s) {                   // s = k*2 + c
        const int k = s >> 1;
        const int c = s & 1;

        // Issue global loads for the next step early (latency hiding).
        if (s + 1 < 54) {
            const uint4* g = (const uint4*)(bpack + (size_t)(s + 1) * STEP_HALVES);
            #pragma unroll
            for (int j = 0; j < U4_PER_THR; ++j) stage[j] = g[tid + j * 256];
        }

        // Gather the A fragment for this (k, c).
        int idx = nbr[(size_t)k * rows + row0 + m];
        unsigned msk  = (idx >= 0) ? 0xFFFFFFFFu : 0u;
        size_t   base = (size_t)(idx >= 0 ? idx : 0) * 64;
        int o0 = c * 32 + (hi ? 8 : 0);
        uint4 alo = *(const uint4*)(xin + base + o0);
        uint4 ahi = *(const uint4*)(xin + base + o0 + 16);
        alo.x &= msk; alo.y &= msk; alo.z &= msk; alo.w &= msk;
        ahi.x &= msk; ahi.y &= msk; ahi.z &= msk; ahi.w &= msk;
        union { uint4 u[2]; v16h h; } au;
        au.u[0] = alo; au.u[1] = ahi;

        // WMMAs for this step; B fragments read from LDS.
        const _Float16* lbuf = ldsb[s & 1];
        #pragma unroll
        for (int nt = 0; nt < NT; ++nt) {
            v16h b = *(const v16h*)(lbuf + (size_t)nt * 512 + lane * 16);
            acc[nt] = __builtin_amdgcn_wmma_f32_16x16x32_f16(
                false, au.h, false, b, (short)0, acc[nt], false, false);
        }

        __syncthreads();                             // step s consumed everywhere
        if (s + 1 < 54) {
            uint4* l = (uint4*)ldsb[(s + 1) & 1];
            #pragma unroll
            for (int j = 0; j < U4_PER_THR; ++j) l[tid + j * 256] = stage[j];
        }
        __syncthreads();                             // step s+1 visible
    }

    // D store: lane<16 -> N=lane, M=v (0..7); lane>=16 -> N=lane-16, M=8+v.
    const int col   = lane & 15;
    const int rbase = row0 + (hi ? 8 : 0);
    #pragma unroll
    for (int nt = 0; nt < NT; ++nt) {
        #pragma unroll
        for (int v = 0; v < 8; ++v) {
            out[(size_t)(rbase + v) * (NT * 16) + nt * 16 + col] = acc[nt][v];
        }
    }
}

// ---------------------------------------------------------------------------
// Host-side launch sequence (graph-capture safe, deterministic).
// ---------------------------------------------------------------------------
extern "C" void kernel_launch(void* const* d_in, const int* in_sizes, int n_in,
                              void* d_out, int out_size, void* d_ws, size_t ws_size,
                              hipStream_t stream) {
    const float* feats    = (const float*)d_in[0];   // [N,3]
    const float* W1       = (const float*)d_in[1];   // [27,3,64]
    const float* W2       = (const float*)d_in[2];   // [27,64,64]
    const float* W3       = (const float*)d_in[3];   // [27,64,128]
    const int*   nbr1     = (const int*)d_in[4];     // [27,N]
    const int*   pool_map = (const int*)d_in[5];     // [8,M]
    const int*   nbr2     = (const int*)d_in[6];     // [27,M]
    float*       outp     = (float*)d_out;           // [M,128]

    char* ws = (char*)d_ws;
    float*     stats1   = (float*)(ws + 0);          // 128 f
    float*     stats2   = (float*)(ws + 512);        // 128 f
    float*     stats3   = (float*)(ws + 1024);       // 256 f
    _Float16*  bpack2   = (_Float16*)(ws + 4096);    // 27*2*4*512 h = 216 KB
    _Float16*  bpack3   = (_Float16*)(ws + 229376);  // 27*2*8*512 h = 432 KB
    float*     partials = (float*)(ws + 786432);     // up to 256 KB
    float*     yA       = (float*)(ws + 1048576);    // [N,64] f32 (y1, reused as y2)
    _Float16*  y1n      = (_Float16*)(ws + 1048576 + 67108864);              // [N,64] f16
    _Float16*  pbuf     = (_Float16*)(ws + 1048576 + 67108864 + 33554432);   // [M,64] f16

    const int NB = 256;  // stats partial blocks (fixed -> deterministic)

    // Weight packing (cheap, re-done every call for determinism).
    {
        int tot2 = 27 * 2 * 4 * 512;
        int tot3 = 27 * 2 * 8 * 512;
        pack_w<<<(tot2 + 255) / 256, 256, 0, stream>>>(W2, bpack2, 64, 64, 4, tot2);
        pack_w<<<(tot3 + 255) / 256, 256, 0, stream>>>(W3, bpack3, 64, 128, 8, tot3);
    }

    // conv1 -> yA (raw, pre-norm)
    conv1_kernel<<<NVOX, 64, 0, stream>>>(feats, W1, nbr1, yA, NVOX);

    // stats over yA, then norm+relu -> f16
    stats_partial<<<NB, 256, 0, stream>>>(yA, NVOX, 64, partials);
    stats_final<<<1, 256, 0, stream>>>(partials, 64, NB, stats1);
    {
        long total = (long)NVOX * 64;
        normrelu_f16<<<4096, 256, 0, stream>>>(yA, y1n, total, 63, stats1, 1.f / NVOX);
    }

    // conv2 (WMMA): 16384 tiles, 8 waves/block -> 2048 blocks; writes yA (reuse)
    conv_wmma<4><<<NVOX / 16 / 8, 256, 0, stream>>>(y1n, bpack2, nbr1, yA, NVOX);

    // stats over conv2 output, then pool (raw) + norm + relu -> f16 pooled
    stats_partial<<<NB, 256, 0, stream>>>(yA, NVOX, 64, partials);
    stats_final<<<1, 256, 0, stream>>>(partials, 64, NB, stats2);
    {
        long total = (long)MVOX * 64;
        pool_normrelu<<<(int)((total + 255) / 256), 256, 0, stream>>>(
            yA, pool_map, stats2, pbuf, MVOX, 1.f / NVOX);
    }

    // conv3 (WMMA, Cout=128): 8192 tiles -> 1024 blocks; raw output into d_out
    conv_wmma<8><<<MVOX / 16 / 8, 256, 0, stream>>>(pbuf, bpack3, nbr2, outp, MVOX);

    // stats over d_out, final in-place norm + relu
    stats_partial<<<NB, 256, 0, stream>>>(outp, MVOX, 128, partials);
    stats_final<<<1, 256, 0, stream>>>(partials, 128, NB, stats3);
    {
        long total = (long)MVOX * 128;
        normrelu_f32<<<4096, 256, 0, stream>>>(outp, total, 127, stats3, 1.f / MVOX);
    }
}
